// SRUppModel_82970178224355
// MI455X (gfx1250) — matrix operand order repaired and checked by reading, not compiled, not run on hardware
//
#include <hip/hip_runtime.h>

// Problem dims (fixed by the reference)
#define BB  32
#define LL  1024
#define DD  256
#define HH  512
#define NCC 16
#define MM  (BB * LL)   // 32768 rows in every big GEMM

typedef __bf16 v16bf __attribute__((ext_vector_type(16)));
typedef __bf16 v8bf  __attribute__((ext_vector_type(8)));
typedef __bf16 bf2   __attribute__((ext_vector_type(2)));
typedef float  v8f   __attribute__((ext_vector_type(8)));

// LDS row stride in bf16 elements: 112 bytes = 7*16B -> every 16B fragment
// chunk is aligned; 28-bank row skew keeps ds_load_b128 conflicts low.
#define LDS_STRIDE 56

__device__ __forceinline__ v16bf frag_cat(v8bf a, v8bf b) {
  return __builtin_shufflevector(a, b, 0, 1, 2, 3, 4, 5, 6, 7,
                                 8, 9, 10, 11, 12, 13, 14, 15);
}

// ---------------------------------------------------------------------------
// GEMM: U[M x N] = A[M x K] * W[K x N], fp32 in/out, computed with split-bf16
// (hi/lo) on v_wmma_f32_16x16x32_bf16 (3 products: hh + lh + hl ~= fp32).
// Workgroup tile 128x128, wave tile 64x32. M,N,K multiples of 128/128/32.
// ---------------------------------------------------------------------------
__global__ __launch_bounds__(256)
void gemm_bf16x2(const float* __restrict__ A, const float* __restrict__ W,
                 float* __restrict__ U, int N, int K) {
  __shared__ __attribute__((aligned(16))) __bf16 Ahi[128 * LDS_STRIDE];
  __shared__ __attribute__((aligned(16))) __bf16 Alo[128 * LDS_STRIDE];
  __shared__ __attribute__((aligned(16))) __bf16 Bhi[128 * LDS_STRIDE];
  __shared__ __attribute__((aligned(16))) __bf16 Blo[128 * LDS_STRIDE];

  const int tid   = threadIdx.x;
  const int nBase = blockIdx.x * 128;
  const int mBase = blockIdx.y * 128;

  const int lane = tid & 31;
  const int w    = tid >> 5;          // wave 0..7
  const int mW   = (w >> 2) * 64;     // wave M offset within block tile
  const int nW   = (w & 3) * 32;      // wave N offset within block tile
  const int half = lane >> 4;         // 0: lanes 0-15, 1: lanes 16-31
  const int lmod = lane & 15;

  const v8f vzero = {0.f, 0.f, 0.f, 0.f, 0.f, 0.f, 0.f, 0.f};
  v8f acc[4][2];
#pragma unroll
  for (int mt = 0; mt < 4; ++mt)
#pragma unroll
    for (int nt = 0; nt < 2; ++nt) acc[mt][nt] = vzero;

  for (int kb = 0; kb < K; kb += 32) {
    // ---- stage A tile (128 x 32) as bf16 hi/lo, layout [m][k], packed k-pairs
    {
      const int k2 = (tid & 15) * 2;   // even k; half-wave covers 128B row span
      const int m0 = tid >> 4;         // 0..15
      const float* src = A + (size_t)(mBase + m0) * K + kb + k2;
#pragma unroll
      for (int r = 0; r < 8; ++r) {
        float v0 = src[(size_t)r * 16 * K];
        float v1 = src[(size_t)r * 16 * K + 1];
        __bf16 h0 = (__bf16)v0;
        __bf16 h1 = (__bf16)v1;
        bf2 hp = {h0, h1};
        bf2 lp = {(__bf16)(v0 - (float)h0), (__bf16)(v1 - (float)h1)};
        int off = (m0 + r * 16) * LDS_STRIDE + k2;     // 4B aligned
        *(bf2*)&Ahi[off] = hp;
        *(bf2*)&Alo[off] = lp;
      }
    }
    // ---- stage B tile (32 x 128) transposed into LDS as [n][k], k-pairs ----
    {
#pragma unroll
      for (int r = 0; r < 8; ++r) {
        int idx = tid + r * 256;       // over 16 k-pairs x 128 n
        int kp  = idx >> 7;            // 0..15
        int n   = idx & 127;           // consecutive lanes -> coalesced global
        int k   = kp * 2;
        float v0 = W[(size_t)(kb + k) * N + nBase + n];
        float v1 = W[(size_t)(kb + k + 1) * N + nBase + n];
        __bf16 h0 = (__bf16)v0;
        __bf16 h1 = (__bf16)v1;
        bf2 hp = {h0, h1};
        bf2 lp = {(__bf16)(v0 - (float)h0), (__bf16)(v1 - (float)h1)};
        int off = n * LDS_STRIDE + k;                  // 4B aligned
        *(bf2*)&Bhi[off] = hp;
        *(bf2*)&Blo[off] = lp;
      }
    }
    __syncthreads();

    // ---- B fragments (ISA layout: n = lane&15, k = (e&7) + 8*half + 16*(e>=8)) ----
    v16bf bh[2], bl[2];
#pragma unroll
    for (int nt = 0; nt < 2; ++nt) {
      const __bf16* pb = &Bhi[(nW + nt * 16 + lmod) * LDS_STRIDE + 8 * half];
      bh[nt] = frag_cat(*(const v8bf*)pb, *(const v8bf*)(pb + 16));
      const __bf16* pl = &Blo[(nW + nt * 16 + lmod) * LDS_STRIDE + 8 * half];
      bl[nt] = frag_cat(*(const v8bf*)pl, *(const v8bf*)(pl + 16));
    }

    // ---- A fragments + 3-product split-bf16 WMMA ----
#pragma unroll
    for (int mt = 0; mt < 4; ++mt) {
      // A layout: m = lane&15, k = (e&7) + 8*half + 16*(e>=8)
      const __bf16* pa = &Ahi[(mW + mt * 16 + lmod) * LDS_STRIDE + 8 * half];
      v16bf ah = frag_cat(*(const v8bf*)pa, *(const v8bf*)(pa + 16));
      const __bf16* pal = &Alo[(mW + mt * 16 + lmod) * LDS_STRIDE + 8 * half];
      v16bf al = frag_cat(*(const v8bf*)pal, *(const v8bf*)(pal + 16));
#pragma unroll
      for (int nt = 0; nt < 2; ++nt) {
        acc[mt][nt] = __builtin_amdgcn_wmma_f32_16x16x32_bf16(
            false, ah, false, bh[nt], (short)0, acc[mt][nt], false, false);
        acc[mt][nt] = __builtin_amdgcn_wmma_f32_16x16x32_bf16(
            false, al, false, bh[nt], (short)0, acc[mt][nt], false, false);
        acc[mt][nt] = __builtin_amdgcn_wmma_f32_16x16x32_bf16(
            false, ah, false, bl[nt], (short)0, acc[mt][nt], false, false);
      }
    }
    __syncthreads();
  }

  // ---- store D (C/D layout: m = i + 8*half, n = lane&15) ----
#pragma unroll
  for (int mt = 0; mt < 4; ++mt)
#pragma unroll
    for (int nt = 0; nt < 2; ++nt) {
      int row0 = mBase + mW + mt * 16 + 8 * half;
      int col  = nBase + nW + nt * 16 + lmod;
#pragma unroll
      for (int i = 0; i < 8; ++i)
        U[(size_t)(row0 + i) * N + col] = acc[mt][nt][i];
    }
}

// ---------------------------------------------------------------------------
// SRU recurrence: one thread per (b, channel); bandwidth-bound, coalesced
// along the channel dimension at every timestep.
// U layout per row: [xt | fpre | rpre | (xres if first)], stride strideU.
// ---------------------------------------------------------------------------
__global__ __launch_bounds__(256)
void sru_scan(const float* __restrict__ U, int strideU,
              const float* __restrict__ hres,  // residual input (ignored if first)
              const float* __restrict__ bias,  // [2H] = bf ++ br
              float* __restrict__ out,         // (B,L,H) highway output
              float* __restrict__ cfin,        // (B,H) final cell state
              int first) {
  int idx = blockIdx.x * 256 + threadIdx.x;   // 0 .. B*H-1
  int b = idx >> 9;                           // /H (H=512)
  int j = idx & (HH - 1);

  const float* Urow = U + (size_t)b * LL * strideU + j;
  const float bf = bias[j];
  const float br = bias[HH + j];
  const size_t hoff = (size_t)b * LL * HH + j;

  float c = 0.f;
  for (int t = 0; t < LL; ++t) {
    const float* u = Urow + (size_t)t * strideU;
    float xt = u[0];
    float f  = 1.f / (1.f + __expf(-(u[HH] + bf)));
    float r  = 1.f / (1.f + __expf(-(u[2 * HH] + br)));
    c = f * c + (1.f - f) * xt;
    float xres = first ? u[3 * HH] : hres[hoff + (size_t)t * HH];
    out[hoff + (size_t)t * HH] = r * c + (1.f - r) * xres;
  }
  cfin[b * HH + j] = c;
}

// ---------------------------------------------------------------------------
// Head: projected = h[:, L-1, :] @ Wp + bp ; output = projected @ Wc + bc.
// Tiny (17 MFLOP) -> one block of plain FMAs with LDS for the intermediate.
// ---------------------------------------------------------------------------
__global__ __launch_bounds__(256)
void head_kernel(const float* __restrict__ h, const float* __restrict__ Wp,
                 const float* __restrict__ bp, const float* __restrict__ Wc,
                 const float* __restrict__ bc, float* __restrict__ out) {
  __shared__ float P[BB * HH];  // 64 KB
  const int tid = threadIdx.x;

#pragma unroll 1
  for (int i = 0; i < (BB * HH) / 256; ++i) {
    int idx = tid + i * 256;
    int b = idx >> 9;
    int j = idx & (HH - 1);
    const float* hl = h + ((size_t)b * LL + (LL - 1)) * HH;
    float s = bp[j];
    for (int k = 0; k < HH; ++k) s += hl[k] * Wp[k * HH + j];
    P[idx] = s;
  }
  __syncthreads();

#pragma unroll 1
  for (int i = 0; i < (BB * NCC) / 256; ++i) {
    int idx = tid + i * 256;
    int b = idx >> 4;
    int n = idx & (NCC - 1);
    float s = bc[n];
    for (int k = 0; k < HH; ++k) s += P[b * HH + k] * Wc[k * NCC + n];
    out[b * NCC + n] = s;
  }
}

// ---------------------------------------------------------------------------
extern "C" void kernel_launch(void* const* d_in, const int* in_sizes, int n_in,
                              void* d_out, int out_size, void* d_ws, size_t ws_size,
                              hipStream_t stream) {
  const float* x  = (const float*)d_in[0];
  const float* W0 = (const float*)d_in[1];
  const float* b0 = (const float*)d_in[2];
  const float* W1 = (const float*)d_in[3];
  const float* b1 = (const float*)d_in[4];
  const float* W2 = (const float*)d_in[5];
  const float* b2 = (const float*)d_in[6];
  const float* W3 = (const float*)d_in[7];
  const float* b3 = (const float*)d_in[8];
  const float* Wp = (const float*)d_in[9];
  const float* bp = (const float*)d_in[10];
  const float* Wc = (const float*)d_in[11];
  const float* bc = (const float*)d_in[12];

  // d_out layout (reference return order): output(B*NC) | h(B*L*H) | hidden(4*B*H)
  float* out_output = (float*)d_out;
  float* out_h      = out_output + BB * NCC;
  float* out_hidden = out_h + (size_t)BB * LL * HH;

  // workspace: U (max B*L*4H f32 = 256MB) | hA (64MB) | hB (64MB)
  float* U  = (float*)d_ws;
  float* hA = U + (size_t)MM * 4 * HH;
  float* hB = hA + (size_t)MM * HH;

  dim3 blk(256);
  dim3 g0((4 * HH) / 128, MM / 128);  // 16 x 256 blocks
  dim3 g1((3 * HH) / 128, MM / 128);  // 12 x 256 blocks
  dim3 gs((BB * HH) / 256);           // 64 blocks

  // Layer 0 (first=1, xres comes from U's 4th slice; hB passed as dummy hres)
  gemm_bf16x2<<<g0, blk, 0, stream>>>(x, W0, U, 4 * HH, DD);
  sru_scan<<<gs, blk, 0, stream>>>(U, 4 * HH, hB, b0, hA, out_hidden + 0 * BB * HH, 1);

  // Layer 1
  gemm_bf16x2<<<g1, blk, 0, stream>>>(hA, W1, U, 3 * HH, HH);
  sru_scan<<<gs, blk, 0, stream>>>(U, 3 * HH, hA, b1, hB, out_hidden + 1 * BB * HH, 0);

  // Layer 2
  gemm_bf16x2<<<g1, blk, 0, stream>>>(hB, W2, U, 3 * HH, HH);
  sru_scan<<<gs, blk, 0, stream>>>(U, 3 * HH, hB, b2, hA, out_hidden + 2 * BB * HH, 0);

  // Layer 3 -> final h straight into d_out
  gemm_bf16x2<<<g1, blk, 0, stream>>>(hA, W3, U, 3 * HH, HH);
  sru_scan<<<gs, blk, 0, stream>>>(U, 3 * HH, hA, b3, out_h, out_hidden + 3 * BB * HH, 0);

  // Head
  head_kernel<<<dim3(1), blk, 0, stream>>>(out_h, Wp, bp, Wc, bc, out_output);
}
